// Cross_Correlation_83580063580168
// MI455X (gfx1250) — matrix operand order repaired and checked by reading, not compiled, not run on hardware
//
#include <hip/hip_runtime.h>
#include <hip/hip_bf16.h>
#include <cstdint>
#include <cstddef>

// ---------------------------------------------------------------------------
// MI455X / gfx1250 implementation: all GEMM-shaped work on v_wmma_f32_16x16x32_f16
// ---------------------------------------------------------------------------

typedef __attribute__((ext_vector_type(16))) _Float16 v16h;
typedef __attribute__((ext_vector_type(8)))  _Float16 v8h;
typedef __attribute__((ext_vector_type(8)))  float    v8f;

#define T_LEN 512
#define BSZ   4
#define NHEAD 8
#define HDIM  64
#define BHN   32            // BSZ*NHEAD
#define DM    512
#define CH    128           // 2*HDIM channels inside temporal block
#define TOPK  6
#define JIN   32
#define KMAX  256           // max adjusted kernel size (T - ind, ind >= T/2)

// ---- WMMA fragment helpers (layouts per CDNA5 ISA 7.12.2, wave32) ----------

__device__ __forceinline__ v16h load16h(const _Float16* p) {
  v8h a = *reinterpret_cast<const v8h*>(p);
  v8h b = *reinterpret_cast<const v8h*>(p + 8);
  v16h r;
#pragma unroll
  for (int i = 0; i < 8; ++i) { r[i] = a[i]; r[i + 8] = b[i]; }
  return r;
}

// A (16x32 f16, row-major source): lane<16 holds K {0..7,16..23}, lane>=16 +8.
__device__ __forceinline__ v16h loadA(const _Float16* row, int kbase, int hi) {
  v8h a = *reinterpret_cast<const v8h*>(row + kbase + hi * 8);
  v8h b = *reinterpret_cast<const v8h*>(row + kbase + hi * 8 + 16);
  v16h r;
#pragma unroll
  for (int i = 0; i < 8; ++i) { r[i] = a[i]; r[i + 8] = b[i]; }
  return r;
}

// B (32x16): lane n = column n == row n of the row-major weight matrix;
// lane<16 holds K {0..7,16..23}, lane>=16 +8 (same order as A).
__device__ __forceinline__ v16h loadB(const _Float16* row, int kbase, int hi) {
  return loadA(row, kbase, hi);
}

__device__ __forceinline__ v8f wmma16(v16h a, v16h b, v8f c) {
  return __builtin_amdgcn_wmma_f32_16x16x32_f16(false, a, false, b, (short)0, c,
                                                false, false);
}

// ---- tiny deterministic RNG (stand-in for jax fold_in/normal) --------------
__device__ __forceinline__ unsigned mixh(unsigned x) {
  x ^= x >> 16; x *= 0x7feb352du; x ^= x >> 15; x *= 0x846ca68bu; x ^= x >> 16;
  return x;
}
__device__ __forceinline__ float u01(unsigned h) {
  return (float)(h >> 8) * (1.0f / 16777216.0f) + (0.5f / 16777216.0f);
}
__device__ __forceinline__ float gaussh(unsigned seed) {
  unsigned h1 = mixh(seed ^ 0x9e3779b9u);
  unsigned h2 = mixh(seed + 0x85ebca6bu);
  return sqrtf(-2.f * __logf(u01(h1))) * __cosf(6.28318530718f * u01(h2));
}

// ---------------------------------------------------------------------------

__global__ void cvt_f16(_Float16* dst, const float* src, int n) {
  int i = blockIdx.x * blockDim.x + threadIdx.x;
  for (; i < n; i += gridDim.x * blockDim.x) dst[i] = (_Float16)src[i];
}

__global__ void zero_f32(float* dst, int n) {
  int i = blockIdx.x * blockDim.x + threadIdx.x;
  for (; i < n; i += gridDim.x * blockDim.x) dst[i] = 0.f;
}

// cosine-sim graph over weights_arr: ji_mat, top-32 idx + softmax weights.
__global__ void graph_kernel(const float* __restrict__ W, float* __restrict__ ji_mat,
                             int* __restrict__ tk_idx, float* __restrict__ tk_w) {
  __shared__ float w[HDIM][JIN];
  __shared__ float nrm[HDIM];
  int tid = threadIdx.x;  // 64 threads, one row each
  for (int idx = tid; idx < HDIM * JIN; idx += HDIM) w[idx >> 5][idx & 31] = W[idx];
  __syncthreads();
  float s = 0.f;
  for (int m = 0; m < JIN; ++m) s += w[tid][m] * w[tid][m];
  nrm[tid] = sqrtf(s);
  __syncthreads();
  float cosr[HDIM];
  float mx = -1e30f;
  for (int j = 0; j < HDIM; ++j) {
    float d = 0.f;
    for (int m = 0; m < JIN; ++m) d += w[tid][m] * w[j][m];
    d = d / (nrm[tid] * nrm[j]);
    cosr[j] = d;
    mx = fmaxf(mx, d);
  }
  float den = 0.f;
  for (int j = 0; j < HDIM; ++j) den += __expf(cosr[j] - mx);
  for (int j = 0; j < HDIM; ++j) ji_mat[tid * HDIM + j] = __expf(cosr[j] - mx) / den;
  // top-32 selection (descending)
  unsigned long long used = 0ull;
  float vals[JIN]; int idxs[JIN];
  for (int m = 0; m < JIN; ++m) {
    float best = -1e30f; int bi = 0;
    for (int j = 0; j < HDIM; ++j) {
      if ((used >> j) & 1ull) continue;
      if (cosr[j] > best) { best = cosr[j]; bi = j; }
    }
    used |= 1ull << bi;
    vals[m] = best; idxs[m] = bi;
  }
  float m2 = vals[0], d2 = 0.f;
  for (int m = 0; m < JIN; ++m) d2 += __expf(vals[m] - m2);
  for (int m = 0; m < JIN; ++m) {
    tk_idx[tid * JIN + m] = idxs[m];
    tk_w[tid * JIN + m]   = __expf(vals[m] - m2) / d2;
  }
}

// X[2048x512] @ W^T + b. mode 0/1: store [bh][d][t] f16. mode 2: store v2[bh][t][d].
__global__ void gemm_qkv(const _Float16* __restrict__ X, const _Float16* __restrict__ Wt,
                         const float* __restrict__ bias, _Float16* __restrict__ dstQK,
                         _Float16* __restrict__ dstV, int mode) {
  int wave = blockIdx.x * (blockDim.x >> 5) + (threadIdx.x >> 5);
  int mt = wave >> 5, nt = wave & 31;          // 128 x 32 tiles
  int m0 = mt * 16, n0 = nt * 16;
  int lane = threadIdx.x & 31, lm = lane & 15, hi = lane >> 4;
  const _Float16* arow = X + (size_t)(m0 + lm) * DM;
  const _Float16* brow = Wt + (size_t)(n0 + lm) * DM;
  v8f acc = {};
  for (int kb = 0; kb < DM; kb += 32) {
    v16h a = loadA(arow, kb, hi);
    v16h b = loadB(brow, kb, hi);
    acc = wmma16(a, b, acc);
  }
#pragma unroll
  for (int r = 0; r < 8; ++r) {
    int m = m0 + r + hi * 8;
    int n = n0 + lm;
    float v = acc[r] + bias[n];
    int t = m >> 2, bb = m & 3;       // rows are (t*B + b)
    int h = n >> 6, d = n & 63;
    int bh = bb * NHEAD + h;
    if (mode == 2) dstV[((size_t)bh * T_LEN + t) * CH + d] = (_Float16)v;
    else           dstQK[((size_t)bh * HDIM + d) * T_LEN + t] = (_Float16)v;
  }
}

// v2[..., 64+d] = weights_arr[d][bh] broadcast over t.
__global__ void fill_v2w(const float* __restrict__ W, _Float16* __restrict__ v2) {
  int i = blockIdx.x * blockDim.x + threadIdx.x;
  int total = BHN * T_LEN * HDIM;
  for (; i < total; i += gridDim.x * blockDim.x) {
    int d = i & 63, t = (i >> 6) & 511, bh = i >> 15;
    v2[((size_t)bh * T_LEN + t) * CH + HDIM + d] = (_Float16)W[d * BHN + bh];
  }
}

// kw[bh][i][s] = sum_m tk_w[i][m] * k[bh][tk_idx[i][m]][s]
__global__ void kw_kernel(const _Float16* __restrict__ k16, const int* __restrict__ tk_idx,
                          const float* __restrict__ tk_w, _Float16* __restrict__ kw16) {
  int bh = blockIdx.x >> 6, i = blockIdx.x & 63;
  int s = threadIdx.x;   // 256 threads, two s each
  float a0 = 0.f, a1 = 0.f;
  for (int m = 0; m < JIN; ++m) {
    int j = tk_idx[i * JIN + m];
    float wv = tk_w[i * JIN + m];
    const _Float16* kr = k16 + ((size_t)bh * HDIM + j) * T_LEN;
    a0 += wv * (float)kr[s];
    a1 += wv * (float)kr[s + 256];
  }
  _Float16* o = kw16 + ((size_t)bh * HDIM + i) * T_LEN;
  o[s] = (_Float16)a0; o[s + 256] = (_Float16)a1;
}

// len_corr[bh][t] = (1/64) sum_i sum_s q[bh][i][(s+t)%T] * kw[bh][i][s]
__global__ void lencorr_kernel(const _Float16* __restrict__ q16,
                               const _Float16* __restrict__ kw16,
                               float* __restrict__ len_corr) {
  int bh = blockIdx.x >> 1;
  int t = ((blockIdx.x & 1) << 8) + threadIdx.x;    // 256 threads
  float acc = 0.f;
  for (int i = 0; i < HDIM; ++i) {
    const _Float16* qr = q16 + ((size_t)bh * HDIM + i) * T_LEN;
    const _Float16* kr = kw16 + ((size_t)bh * HDIM + i) * T_LEN;
    for (int s2 = 0; s2 < T_LEN; s2 += 4) {
#pragma unroll
      for (int u = 0; u < 4; ++u) {
        int s = s2 + u;
        acc += (float)qr[(s + t) & (T_LEN - 1)] * (float)kr[s];
      }
    }
  }
  len_corr[bh * T_LEN + t] = acc * (1.f / 64.f);
}

// mean over bh, top-6 over t, softmax; store raw idx [0..5], adjusted ind [6..11], weights.
__global__ void topk_kernel(const float* __restrict__ len_corr, int* __restrict__ ctrl_i,
                            float* __restrict__ ctrl_f) {
  __shared__ float mc[T_LEN];
  int t = threadIdx.x;   // 512 threads
  float s = 0.f;
  for (int bh = 0; bh < BHN; ++bh) s += len_corr[bh * T_LEN + t];
  mc[t] = s * (1.f / 32.f);
  __syncthreads();
  if (t == 0) {
    int raw[TOPK]; float vals[TOPK];
    for (int n = 0; n < TOPK; ++n) {
      float best = -1e30f; int bi = 0;
      for (int tt = 0; tt < T_LEN; ++tt)
        if (mc[tt] > best) { best = mc[tt]; bi = tt; }
      vals[n] = best; raw[n] = bi; mc[bi] = -1e30f;
    }
    float mx = vals[0], den = 0.f;
    for (int n = 0; n < TOPK; ++n) den += __expf(vals[n] - mx);
    for (int n = 0; n < TOPK; ++n) {
      ctrl_f[n] = __expf(vals[n] - mx) / den;
      ctrl_i[n] = raw[n];
      int ind = raw[n];
      if (ind >= T_LEN / 2) ind = T_LEN - ind;
      ctrl_i[TOPK + n] = ind;
    }
  }
}

// qt[bh][i][s] = sum_n tmp_corr[n] * q[bh][i][(s+idx_raw[n])%T]
__global__ void qt_kernel(const _Float16* __restrict__ q16, const int* __restrict__ ctrl_i,
                          const float* __restrict__ ctrl_f, _Float16* __restrict__ qt16) {
  int bh = blockIdx.x >> 6, i = blockIdx.x & 63;
  int s = threadIdx.x;
  const _Float16* qr = q16 + ((size_t)bh * HDIM + i) * T_LEN;
  float a0 = 0.f, a1 = 0.f;
  for (int n = 0; n < TOPK; ++n) {
    int sh = ctrl_i[n]; float tc = ctrl_f[n];
    a0 += tc * (float)qr[(s + sh) & (T_LEN - 1)];
    a1 += tc * (float)qr[(s + 256 + sh) & (T_LEN - 1)];
  }
  _Float16* o = qt16 + ((size_t)bh * HDIM + i) * T_LEN;
  o[s] = (_Float16)a0; o[s + 256] = (_Float16)a1;
}

// dim_cor_raw[i][j] = (1/32) sum_b sum_s qt[b][i][s] * k[b][j][s]  (64x64, K=16384)
__global__ void dimcor_gemm(const _Float16* __restrict__ qt16,
                            const _Float16* __restrict__ k16, float* __restrict__ dim_raw) {
  int wave = threadIdx.x >> 5;                 // 16 waves in one block
  int i0 = (wave >> 2) << 4, j0 = (wave & 3) << 4;
  int lane = threadIdx.x & 31, lm = lane & 15, hi = lane >> 4;
  v8f acc = {};
  for (int b = 0; b < BHN; ++b) {
    const _Float16* arow = qt16 + ((size_t)b * HDIM + i0 + lm) * T_LEN;
    const _Float16* brow = k16  + ((size_t)b * HDIM + j0 + lm) * T_LEN;
    for (int s = 0; s < T_LEN; s += 32) {
      v16h a  = loadA(arow, s, hi);
      v16h bb = loadB(brow, s, hi);
      acc = wmma16(a, bb, acc);
    }
  }
#pragma unroll
  for (int r = 0; r < 8; ++r)
    dim_raw[(i0 + r + hi * 8) * HDIM + j0 + lm] = acc[r] * (1.f / 32.f);
}

__global__ void dimcor_softmax(const float* __restrict__ dim_raw, float* __restrict__ out) {
  int i = threadIdx.x;   // 64 threads
  float mx = -1e30f;
  for (int j = 0; j < HDIM; ++j) mx = fmaxf(mx, 0.125f * dim_raw[i * HDIM + j]);
  float den = 0.f;
  for (int j = 0; j < HDIM; ++j) den += __expf(0.125f * dim_raw[i * HDIM + j] - mx);
  for (int j = 0; j < HDIM; ++j)
    out[i * HDIM + j] = __expf(0.125f * dim_raw[i * HDIM + j] - mx) / den;
}

// deterministic-per-kernel-size conv weights [k][co][ci] f16 + uniform biases.
__global__ void tb_genw(const int* __restrict__ ind_p, _Float16* __restrict__ w1,
                        _Float16* __restrict__ w2, float* __restrict__ b1,
                        float* __restrict__ b2) {
  int ind = *ind_p;
  if (ind <= 1) return;
  if (blockIdx.x == KMAX * CH) {
    int ci = threadIdx.x;
    float bound = rsqrtf((float)(CH * ind));
    b1[ci] = (2.f * u01(mixh((unsigned)ind * 1315423911u + ci * 2654435761u + 17u)) - 1.f) * bound;
    b2[ci] = (2.f * u01(mixh((unsigned)ind * 1315423911u + ci * 2654435761u + 91u)) - 1.f) * bound;
    return;
  }
  int k = blockIdx.x >> 7, co = blockIdx.x & 127, ci = threadIdx.x;
  if (k >= ind) return;
  unsigned base = ((unsigned)ind * 131u + k) * 131u + co;
  base = base * 131u + ci;
  w1[((size_t)k * CH + co) * CH + ci] = (_Float16)(0.01f * gaussh(base * 2u + 1u));
  w2[((size_t)k * CH + co) * CH + ci] = (_Float16)(0.01f * gaussh(base * 2654435761u + 7u));
}

// causal conv1d (C=128, data-dependent K) as Sum_k of 16x16 WMMA tiles.
// pass 1: out = relu(conv(x)+b);  pass 2: out = relu(relu(conv(x)+b) + v2)
__global__ void tb_conv(const int* __restrict__ ind_p, const _Float16* __restrict__ x,
                        const _Float16* __restrict__ wbuf, const float* __restrict__ bias,
                        const _Float16* __restrict__ v2, _Float16* __restrict__ out,
                        int pass) {
  int ind = *ind_p;
  if (ind <= 1) return;
  int wave = blockIdx.x * (blockDim.x >> 5) + (threadIdx.x >> 5);
  int rowT = wave >> 3, co0 = (wave & 7) << 4;     // 1024 row-tiles x 8 co-tiles
  int bh = rowT >> 5, t0 = (rowT & 31) << 4;
  int lane = threadIdx.x & 31, lm = lane & 15, hi = lane >> 4;
  int tl = t0 + lm;
  v8f acc = {};
  for (int k = 0; k < ind; ++k) {
    int tp = tl + k - (ind - 1);                   // causal left pad
    const _Float16* xr = x + ((size_t)bh * T_LEN + tp) * CH;
    const _Float16* wk = wbuf + (size_t)k * CH * CH + (size_t)(co0 + lm) * CH;
    bool valid = tp >= 0;
#pragma unroll
    for (int cc = 0; cc < 4; ++cc) {
      v16h a = {};
      if (valid) a = loadA(xr, cc * 32, hi);
      v16h b = loadB(wk, cc * 32, hi);
      acc = wmma16(a, b, acc);
    }
  }
#pragma unroll
  for (int r = 0; r < 8; ++r) {
    int t = t0 + r + hi * 8, co = co0 + lm;
    size_t o = ((size_t)bh * T_LEN + t) * CH + co;
    float v = fmaxf(acc[r] + bias[co], 0.f);
    if (pass == 2) v = fmaxf(v + (float)v2[o], 0.f);
    out[o] = (_Float16)v;
  }
}

__global__ void tb_accum(const int* __restrict__ ind_p, const float* __restrict__ tc_p,
                         const _Float16* __restrict__ y, const _Float16* __restrict__ v2,
                         float* __restrict__ agg) {
  int ind = *ind_p; float tc = *tc_p;
  const _Float16* src = (ind <= 1) ? v2 : y;
  int i = blockIdx.x * blockDim.x + threadIdx.x;
  int total = BHN * T_LEN * CH;
  for (; i < total; i += gridDim.x * blockDim.x) agg[i] += tc * (float)src[i];
}

// out[t*B+b][n] = sum_kk agg16[(b*8 + kk/128)][t][kk%128] * Wo[n][kk] + bo[n]
__global__ void gemm_out(const _Float16* __restrict__ agg16, const _Float16* __restrict__ Wo16,
                         const float* __restrict__ bo, float* __restrict__ out) {
  int wave = blockIdx.x * (blockDim.x >> 5) + (threadIdx.x >> 5);
  int mt = wave >> 5, nt = wave & 31;
  int m0 = mt * 16, n0 = nt * 16;
  int lane = threadIdx.x & 31, lm = lane & 15, hi = lane >> 4;
  int m = m0 + lm;
  int t = m >> 2, bb = m & 3;
  const _Float16* brow = Wo16 + (size_t)(n0 + lm) * (2 * DM);
  v8f acc = {};
  for (int kb = 0; kb < 2 * DM; kb += 32) {
    int h = kb >> 7, c = kb & 127;
    const _Float16* arow = agg16 + ((size_t)(bb * NHEAD + h) * T_LEN + t) * CH + c;
    v16h a = loadA(arow, 0, hi);
    v16h b = loadB(brow, kb, hi);
    acc = wmma16(a, b, acc);
  }
#pragma unroll
  for (int r = 0; r < 8; ++r) {
    int mm = m0 + r + hi * 8, n = n0 + lm;
    out[(size_t)mm * DM + n] = acc[r] + bo[n];
  }
}

// ---------------------------------------------------------------------------

extern "C" void kernel_launch(void* const* d_in, const int* in_sizes, int n_in,
                              void* d_out, int out_size, void* d_ws, size_t ws_size,
                              hipStream_t stream) {
  (void)in_sizes; (void)n_in; (void)out_size; (void)ws_size;
  const float* query = (const float*)d_in[0];
  const float* key_in = (const float*)d_in[1];
  const float* value = (const float*)d_in[2];
  /* d_in[3] attn_mask unused by reference */
  const float* Wq = (const float*)d_in[4];
  const float* bq = (const float*)d_in[5];
  const float* Wk = (const float*)d_in[6];
  const float* bk = (const float*)d_in[7];
  const float* Wv = (const float*)d_in[8];
  const float* bv = (const float*)d_in[9];
  const float* Wo = (const float*)d_in[10];
  const float* bo = (const float*)d_in[11];
  const float* Warr = (const float*)d_in[12];

  float* out_main = (float*)d_out;                 // [512,4,512]
  float* dim_cor_out = out_main + 512 * 4 * 512;   // [64,64]
  float* ji_out = dim_cor_out + 64 * 64;           // [64,64]

  char* base = (char*)d_ws;
  size_t off = 0;
  auto alloc = [&](size_t bytes) -> void* {
    void* p = base + off;
    off += (bytes + 255) & ~(size_t)255;
    return p;
  };
  _Float16* xq16 = (_Float16*)alloc((size_t)2048 * 512 * 2);
  _Float16* xk16 = (_Float16*)alloc((size_t)2048 * 512 * 2);
  _Float16* xv16 = (_Float16*)alloc((size_t)2048 * 512 * 2);
  _Float16* Wq16 = (_Float16*)alloc((size_t)512 * 512 * 2);
  _Float16* Wk16 = (_Float16*)alloc((size_t)512 * 512 * 2);
  _Float16* Wv16 = (_Float16*)alloc((size_t)512 * 512 * 2);
  _Float16* Wo16 = (_Float16*)alloc((size_t)512 * 1024 * 2);
  _Float16* q16  = (_Float16*)alloc((size_t)BHN * HDIM * T_LEN * 2);
  _Float16* k16  = (_Float16*)alloc((size_t)BHN * HDIM * T_LEN * 2);
  _Float16* v2_16 = (_Float16*)alloc((size_t)BHN * T_LEN * CH * 2);
  _Float16* kw16 = (_Float16*)alloc((size_t)BHN * HDIM * T_LEN * 2);
  _Float16* qt16 = (_Float16*)alloc((size_t)BHN * HDIM * T_LEN * 2);
  float* len_corr = (float*)alloc((size_t)BHN * T_LEN * 4);
  int*   tk_idx   = (int*)alloc((size_t)HDIM * JIN * 4);
  float* tk_w     = (float*)alloc((size_t)HDIM * JIN * 4);
  int*   ctrl_i   = (int*)alloc(64 * 4);
  float* ctrl_f   = (float*)alloc(64 * 4);
  float* dim_raw  = (float*)alloc((size_t)HDIM * HDIM * 4);
  _Float16* w1buf = (_Float16*)alloc((size_t)KMAX * CH * CH * 2);
  _Float16* w2buf = (_Float16*)alloc((size_t)KMAX * CH * CH * 2);
  float* b1 = (float*)alloc(CH * 4);
  float* b2 = (float*)alloc(CH * 4);
  _Float16* h16 = (_Float16*)alloc((size_t)BHN * T_LEN * CH * 2);
  _Float16* y16 = (_Float16*)alloc((size_t)BHN * T_LEN * CH * 2);
  float* agg = (float*)alloc((size_t)BHN * T_LEN * CH * 4);
  _Float16* agg16 = (_Float16*)alloc((size_t)BHN * T_LEN * CH * 2);

  // --- f32 -> f16 staging ---
  cvt_f16<<<512, 256, 0, stream>>>(xq16, query, 2048 * 512);
  cvt_f16<<<512, 256, 0, stream>>>(xk16, key_in, 2048 * 512);
  cvt_f16<<<512, 256, 0, stream>>>(xv16, value, 2048 * 512);
  cvt_f16<<<256, 256, 0, stream>>>(Wq16, Wq, 512 * 512);
  cvt_f16<<<256, 256, 0, stream>>>(Wk16, Wk, 512 * 512);
  cvt_f16<<<256, 256, 0, stream>>>(Wv16, Wv, 512 * 512);
  cvt_f16<<<512, 256, 0, stream>>>(Wo16, Wo, 512 * 1024);

  // --- cosine graph / top-32 ---
  graph_kernel<<<1, 64, 0, stream>>>(Warr, ji_out, tk_idx, tk_w);

  // --- Q/K/V projections (WMMA) ---
  gemm_qkv<<<1024, 128, 0, stream>>>(xq16, Wq16, bq, q16, v2_16, 0);
  gemm_qkv<<<1024, 128, 0, stream>>>(xk16, Wk16, bk, k16, v2_16, 1);
  gemm_qkv<<<1024, 128, 0, stream>>>(xv16, Wv16, bv, q16, v2_16, 2);
  fill_v2w<<<1024, 256, 0, stream>>>(Warr, v2_16);

  // --- contracted cross-correlation -> top-6 time shifts ---
  kw_kernel<<<BHN * HDIM, 256, 0, stream>>>(k16, tk_idx, tk_w, kw16);
  lencorr_kernel<<<64, 256, 0, stream>>>(q16, kw16, len_corr);
  topk_kernel<<<1, 512, 0, stream>>>(len_corr, ctrl_i, ctrl_f);

  // --- dim_cor via shift-combined GEMM (WMMA) ---
  qt_kernel<<<BHN * HDIM, 256, 0, stream>>>(q16, ctrl_i, ctrl_f, qt16);
  dimcor_gemm<<<1, 512, 0, stream>>>(qt16, k16, dim_raw);
  dimcor_softmax<<<1, 64, 0, stream>>>(dim_raw, dim_cor_out);

  // --- temporal-block aggregation over top-6 shifts (WMMA convs) ---
  zero_f32<<<1024, 256, 0, stream>>>(agg, BHN * T_LEN * CH);
  for (int n = 0; n < TOPK; ++n) {
    const int* indp = ctrl_i + TOPK + n;
    tb_genw<<<KMAX * CH + 1, 128, 0, stream>>>(indp, w1buf, w2buf, b1, b2);
    tb_conv<<<1024, 256, 0, stream>>>(indp, v2_16, w1buf, b1, v2_16, h16, 1);
    tb_conv<<<1024, 256, 0, stream>>>(indp, h16, w2buf, b2, v2_16, y16, 2);
    tb_accum<<<1024, 256, 0, stream>>>(indp, ctrl_f + n, y16, v2_16, agg);
  }

  // --- output projection (WMMA) ---
  cvt_f16<<<512, 256, 0, stream>>>(agg16, agg, BHN * T_LEN * CH);
  gemm_out<<<1024, 128, 0, stream>>>(agg16, Wo16, bo, out_main);
}
